// AlignmentNetwork_5162550690100
// MI455X (gfx1250) — matrix operand order repaired and checked by reading, not compiled, not run on hardware
//
#include <hip/hip_runtime.h>
#include <hip/hip_bf16.h>
#include <math.h>

typedef __attribute__((ext_vector_type(16))) _Float16 v16h;
typedef __attribute__((ext_vector_type(8)))  float    v8f;
typedef unsigned int u32x4 __attribute__((ext_vector_type(4)));
typedef int          i32x4 __attribute__((ext_vector_type(4)));
typedef int          i32x8 __attribute__((ext_vector_type(8)));

#define TEMPERATURE 0.0005f
#define B_   8
#define CQ   80
#define CK   512
#define CA   80
#define TDE  800
#define TEN  160
#define CK2  1024
#define CQ2  160

// TDM builtins exist only in the DEVICE compilation pass (host pass lacks them).
#if defined(__HIP_DEVICE_COMPILE__)
#if !__has_builtin(__builtin_amdgcn_tensor_load_to_lds) || !__has_builtin(__builtin_amdgcn_s_wait_tensorcnt)
#error "tensor_load_to_lds builtin not available in gfx1250 device pass"
#endif
#endif

#define WMMA_F16(A, Bv, C) \
  __builtin_amdgcn_wmma_f32_16x16x32_f16(false, (A), false, (Bv), (short)0, (C), false, false)

union V16U { v16h v; uint4 u[2]; };

// ------------- linear fragment loaders: per-lane base pointer + kc-linear offset -------------
__device__ __forceinline__ v16h load_a16(const _Float16* __restrict__ p, int kc) {
  V16U r;
  r.u[0] = *(const uint4*)(p + kc);
  r.u[1] = *(const uint4*)(p + kc + 16);
  return r.v;
}
__device__ __forceinline__ v16h load_b16(const _Float16* __restrict__ p, int kc) {
  V16U r;
  r.u[0] = *(const uint4*)(p + kc);
  r.u[1] = *(const uint4*)(p + kc + 16);
  return r.v;
}

// C/D fragment (8 f32): lanes 0-15: N=lane, M=r; lanes 16-31: N=lane-16, M=r+8.
__device__ __forceinline__ void store_c_rm_lds(_Float16* lds, const v8f& c, int m0, int pitch,
                                               const float* __restrict__ bias, bool relu) {
  const int lane = threadIdx.x & 31;
  const int n    = lane & 15;
  const int mo   = (lane < 16) ? 0 : 8;
  union { _Float16 h[8]; uint4 u; } t;
#pragma unroll
  for (int r = 0; r < 8; ++r) {
    float v = c[r] + bias[m0 + mo + r];
    if (relu) v = (v > 0.0f) ? v : 0.0f;
    t.h[r] = (_Float16)v;
  }
  *(uint4*)(lds + (long)n * pitch + m0 + mo) = t.u;
}

__device__ __forceinline__ void store_c_bias_global(float* __restrict__ out, const v8f& c,
                                                    int m0, int n0, const float* __restrict__ bias,
                                                    int m_stride) {
  const int lane = threadIdx.x & 31;
  const int n    = n0 + (lane & 15);
  const int mo   = (lane < 16) ? 0 : 8;
#pragma unroll
  for (int r = 0; r < 8; ++r) {
    const int m = m0 + mo + r;
    out[(long)m * m_stride + n] = c[r] + bias[m];
  }
}

// ------------- Tensor Data Mover: 2-D f16 tile global->LDS (ISA ch.8 D# layout) -------------
__device__ __forceinline__ void tdm_load_2d_f16(unsigned lds_addr, const void* gaddr,
                                                unsigned tensor_d0, unsigned tensor_d1,
                                                unsigned tile_d0, unsigned tile_d1,
                                                unsigned d0_stride) {
#if defined(__HIP_DEVICE_COMPILE__)
  const unsigned long ga = (unsigned long)gaddr;
  u32x4 g0;
  g0[0] = 1u;                                                   // count=1, user mode
  g0[1] = lds_addr;                                             // lds_addr [63:32]
  g0[2] = (unsigned)ga;                                         // global_addr [95:64]
  g0[3] = (unsigned)((ga >> 32) & 0x01FFFFFFu) | (2u << 30);    // global_addr hi + type=2
  i32x8 g1;
  g1[0] = (int)(1u << 16);                                      // data_size=1 (2 bytes)
  g1[1] = (int)((tensor_d0 & 0xFFFFu) << 16);                   // tensor_dim0 [79:48]
  g1[2] = (int)((tensor_d0 >> 16) | ((tensor_d1 & 0xFFFFu) << 16));
  g1[3] = (int)((tensor_d1 >> 16) | (tile_d0 << 16));           // | tile_dim0 [127:112]
  g1[4] = (int)(tile_d1 & 0xFFFFu);                             // tile_dim1 (tile_dim2=0)
  g1[5] = (int)d0_stride;                                       // tensor_dim0_stride lo32
  g1[6] = 0;
  g1[7] = 0;
  i32x4 z4 = {0, 0, 0, 0};
#if defined(__clang_major__) && __clang_major__ >= 23
  i32x8 z8 = {0, 0, 0, 0, 0, 0, 0, 0};
  __builtin_amdgcn_tensor_load_to_lds(g0, g1, z4, z4, z8, 0);
#else
  __builtin_amdgcn_tensor_load_to_lds(g0, g1, z4, z4, 0);
#endif
#else
  (void)lds_addr; (void)gaddr; (void)tensor_d0; (void)tensor_d1;
  (void)tile_d0; (void)tile_d1; (void)d0_stride;
#endif
}

__device__ __forceinline__ void tdm_wait0() {
#if defined(__HIP_DEVICE_COMPILE__)
  __builtin_amdgcn_s_wait_tensorcnt(0);
#endif
}

// ---------------- pack kernels (one-shot, memory-bound) ----------------
__global__ void align_pack_w_kernel(const float* __restrict__ src, _Float16* __restrict__ dst,
                                    int M, int row_stride, int KI, int KIvalid, int ntap,
                                    int col_stride, int tap_stride) {
  const long idx   = (long)blockIdx.x * 256 + threadIdx.x;
  const int  Kdst  = KI * ntap;
  const long total = (long)M * Kdst;
  if (idx >= total) return;
  const int m   = (int)(idx / Kdst);
  const int k   = (int)(idx % Kdst);
  const int tap = k / KI;
  const int ci  = k % KI;
  float v = 0.0f;
  if (ci < KIvalid) v = src[(long)m * row_stride + (long)ci * col_stride + (long)tap * tap_stride];
  dst[idx] = (_Float16)v;
}

__global__ void align_transpose_pad_kernel(const float* __restrict__ src, _Float16* __restrict__ dst,
                                           int C, int T, int Kp, int Tp, int t_off) {
  const long idx   = (long)blockIdx.x * 256 + threadIdx.x;
  const long total = (long)B_ * Tp * Kp;
  if (idx >= total) return;
  const int  c   = (int)(idx % Kp);
  const long rem = idx / Kp;
  const int  t   = (int)(rem % Tp);
  const int  b   = (int)(rem / Tp);
  const int  ts  = t - t_off;
  float v = 0.0f;
  if (c < C && ts >= 0 && ts < T) v = src[((long)b * C + c) * T + ts];
  dst[idx] = (_Float16)v;
}

// ---------------- Kernel: key network (conv3 -> relu -> 1x1) fused in LDS ----------------
#define KINP 512
#define KMIDP 1040
__global__ __launch_bounds__(512) void align_key_net_kernel(
    const _Float16* __restrict__ kw1p, const float* __restrict__ kb1,
    const _Float16* __restrict__ kw2p, const float* __restrict__ kb2,
    const _Float16* __restrict__ keysT, float* __restrict__ key_out) {
  const int b     = blockIdx.x / 10;
  const int tt    = blockIdx.x % 10;
  const int tbase = tt * 16;
  const int wave  = threadIdx.x >> 5;
  const int lane  = threadIdx.x & 31;
  const int l15   = lane & 15;
  const int koff  = (lane < 16) ? 0 : 8;
  const int kh    = (lane >> 4) << 3;
  __shared__ _Float16 kin[18 * KINP];     // keysT rows [tbase .. tbase+17] via TDM
  __shared__ _Float16 mid[16 * KMIDP];    // conv1 tile, row n, col co
  const _Float16* kT = keysT + (long)b * 162 * KINP;

  if (wave == 0) {
    tdm_load_2d_f16((unsigned)(unsigned long)(void*)kin, kT + (long)tbase * KINP,
                    KINP, (unsigned)(162 - tbase), KINP, 18, KINP);
    tdm_wait0();
  }
  __syncthreads();

  // conv1: 4-way register-blocked over M — one B fragment feeds 4 WMMAs.
  // B addr LINEAR in kc: n*512 + kc (halo rows make im2col contiguous).
  {
    const _Float16* pb  = kin + l15 * KINP + kh;
    const int m0 = wave * 64;
    const _Float16* pa0 = kw1p + (long)(m0 + l15) * 1536 + koff;
    v8f acc0 = {}, acc1 = {}, acc2 = {}, acc3 = {};
#pragma unroll 1
    for (int kc = 0; kc < 1536; kc += 32) {
      v16h bf = load_b16(pb, kc);
      acc0 = WMMA_F16(load_a16(pa0, kc),             bf, acc0);
      acc1 = WMMA_F16(load_a16(pa0 + 16 * 1536, kc), bf, acc1);
      acc2 = WMMA_F16(load_a16(pa0 + 32 * 1536, kc), bf, acc2);
      acc3 = WMMA_F16(load_a16(pa0 + 48 * 1536, kc), bf, acc3);
    }
    store_c_rm_lds(mid, acc0, m0,      KMIDP, kb1, true);
    store_c_rm_lds(mid, acc1, m0 + 16, KMIDP, kb1, true);
    store_c_rm_lds(mid, acc2, m0 + 32, KMIDP, kb1, true);
    store_c_rm_lds(mid, acc3, m0 + 48, KMIDP, kb1, true);
  }
  __syncthreads();

  // conv2 (1x1): M=80, K=1024 from LDS (B addr linear: n*KMIDP + kc)
  if (wave < 5) {
    const int m0 = wave * 16;
    const _Float16* pa  = kw2p + (long)(m0 + l15) * CK2 + koff;
    const _Float16* pb2 = mid + l15 * KMIDP + kh;
    v8f acc = {};
#pragma unroll 1
    for (int kc = 0; kc < CK2; kc += 32)
      acc = WMMA_F16(load_a16(pa, kc), load_b16(pb2, kc), acc);
    store_c_bias_global(key_out + (long)b * CA * TEN, acc, m0, tbase, kb2, TEN);
  }
}

// ---------------- Kernel: query network (conv3 -> relu -> 1x1 -> relu -> 1x1) ----------------
#define M1P 176
#define M2P 112
__global__ __launch_bounds__(320) void align_query_net_kernel(
    const _Float16* __restrict__ qw1p, const float* __restrict__ qb1,
    const _Float16* __restrict__ qw2p, const float* __restrict__ qb2,
    const _Float16* __restrict__ qw3p, const float* __restrict__ qb3,
    const _Float16* __restrict__ queriesT, float* __restrict__ q_out) {
  const int b     = blockIdx.x / 50;
  const int tt    = blockIdx.x % 50;
  const int tbase = tt * 16;
  const int wave  = threadIdx.x >> 5;
  const int lane  = threadIdx.x & 31;
  const int l15   = lane & 15;
  const int koff  = (lane < 16) ? 0 : 8;
  const int kh    = (lane >> 4) << 3;
  __shared__ _Float16 qin[18 * 96];       // queriesT rows [tbase .. tbase+17] via TDM
  __shared__ _Float16 mid1[16 * M1P];
  __shared__ _Float16 mid2[16 * M2P];
  const _Float16* qT = queriesT + (long)b * 802 * 96;

  if (wave == 0) {
    tdm_load_2d_f16((unsigned)(unsigned long)(void*)qin, qT + (long)tbase * 96,
                    96, (unsigned)(802 - tbase), 96, 18, 96);
    tdm_wait0();
  }
  __syncthreads();

  // conv1: M=160 (one tile per wave), K'=288; B addr linear: n*96 + kc
  {
    const int m0 = wave * 16;
    const _Float16* pa = qw1p + (long)(m0 + l15) * 288 + koff;
    const _Float16* pb = qin + l15 * 96 + kh;
    v8f acc = {};
#pragma unroll
    for (int kc = 0; kc < 288; kc += 32)
      acc = WMMA_F16(load_a16(pa, kc), load_b16(pb, kc), acc);
    store_c_rm_lds(mid1, acc, m0, M1P, qb1, true);
  }
  __syncthreads();

  // conv2 (1x1): M=80, K=160
  if (wave < 5) {
    const int m0 = wave * 16;
    const _Float16* pa = qw2p + (long)(m0 + l15) * CQ2 + koff;
    const _Float16* pb = mid1 + l15 * M1P + kh;
    v8f acc = {};
#pragma unroll
    for (int kc = 0; kc < CQ2; kc += 32)
      acc = WMMA_F16(load_a16(pa, kc), load_b16(pb, kc), acc);
    store_c_rm_lds(mid2, acc, m0, M2P, qb2, true);
  }
  if (threadIdx.x < 16) {                 // zero K-pad cols 80..95 of mid2
    const uint4 z = {0, 0, 0, 0};
    *(uint4*)(mid2 + threadIdx.x * M2P + 80) = z;
    *(uint4*)(mid2 + threadIdx.x * M2P + 88) = z;
  }
  __syncthreads();

  // conv3 (1x1): M=80, K=96 (padded)
  if (wave < 5) {
    const int m0 = wave * 16;
    const _Float16* pa = qw3p + (long)(m0 + l15) * 96 + koff;
    const _Float16* pb = mid2 + l15 * M2P + kh;
    v8f acc = {};
#pragma unroll
    for (int kc = 0; kc < 96; kc += 32)
      acc = WMMA_F16(load_a16(pa, kc), load_b16(pb, kc), acc);
    store_c_bias_global(q_out + (long)b * CA * TDE, acc, m0, tbase, qb3, TDE);
  }
}

// ---------------- Kernel: fused distance GEMM + log_softmax + prior + mask + softmax ----------
__device__ __forceinline__ float block_reduce_160(float v, float* red, int do_sum) {
  const int t = threadIdx.x;
#pragma unroll
  for (int o = 16; o > 0; o >>= 1) {
    float other = __shfl_xor(v, o, 32);
    v = do_sum ? (v + other) : fmaxf(v, other);
  }
  if ((t & 31) == 0) red[t >> 5] = v;
  __syncthreads();
  float r;
  if (do_sum) r = red[0] + red[1] + red[2] + red[3] + red[4];
  else        r = fmaxf(fmaxf(fmaxf(red[0], red[1]), fmaxf(red[2], red[3])), red[4]);
  __syncthreads();
  return r;
}

__global__ __launch_bounds__(160) void align_attn_kernel(
    const _Float16* __restrict__ q_outT, const _Float16* __restrict__ key_outT,
    const float* __restrict__ q_out, const float* __restrict__ key_out,
    const float* __restrict__ attn_prior, const unsigned char* __restrict__ mask,
    float* __restrict__ out) {
  const int b    = blockIdx.x / 50;
  const int mt   = blockIdx.x % 50;
  const int m0   = mt * 16;
  const int t    = threadIdx.x;
  const int wave = t >> 5;
  const int lane = t & 31;
  const int l15  = lane & 15;
  const int koff = (lane < 16) ? 0 : 8;
  const int kh   = (lane >> 4) << 3;
  __shared__ _Float16 kst[TEN * 96];      // whole per-batch key_outT slab via TDM (30 KB)
  __shared__ float Stile[16 * TEN];
  __shared__ float q2s[16];
  __shared__ float k2s[TEN];
  __shared__ float red[8];
  const _Float16* qTa = q_outT   + (long)b * TDE * 96;
  const _Float16* kTb = key_outT + (long)b * TEN * 96;
  const float*    qp  = q_out    + (long)b * CA * TDE;
  const float*    kp  = key_out  + (long)b * CA * TEN;

  if (wave == 0) {
    tdm_load_2d_f16((unsigned)(unsigned long)(void*)kst, kTb, 96, TEN, 96, TEN, 96);
  }
  {
    float s = 0.0f;
    for (int c = 0; c < CA; ++c) { float v = kp[c * TEN + t]; s += v * v; }
    k2s[t] = s;
    if (t < 16) {
      float s2 = 0.0f;
      for (int c = 0; c < CA; ++c) { float v = qp[(long)c * TDE + m0 + t]; s2 += v * v; }
      q2s[t] = s2;
    }
  }
  if (wave == 0) tdm_wait0();
  __syncthreads();

  // q.k tile [16 x 160] via WMMA, K=96 (padded); B served from LDS
  const _Float16* pa = qTa + (long)(m0 + l15) * 96 + koff;
#pragma unroll 1
  for (int nt = wave; nt < 10; nt += 5) {
    const int n0 = nt * 16;
    const _Float16* pb = kst + (n0 + l15) * 96 + kh;
    v8f acc = {};
#pragma unroll
    for (int kc = 0; kc < 96; kc += 32)
      acc = WMMA_F16(load_a16(pa, kc), load_b16(pb, kc), acc);
    const int n  = n0 + l15;
    const int mo = (lane < 16) ? 0 : 8;
#pragma unroll
    for (int r = 0; r < 8; ++r) Stile[(mo + r) * TEN + n] = acc[r];
  }
  __syncthreads();

  const float* prow = attn_prior + ((long)b * TDE + m0) * TEN;
  const unsigned char mk = mask[(long)b * TEN + t];
  const long ATTN_N = (long)B_ * TDE * TEN;

#pragma unroll 1
  for (int r = 0; r < 16; ++r) {
    float logp = -TEMPERATURE * (q2s[r] + k2s[t] - 2.0f * Stile[r * TEN + t]);
    float mx = block_reduce_160(logp, red, 0);
    float se = block_reduce_160(__expf(logp - mx), red, 1);
    logp = (logp - mx) - __logf(se);
    logp += __logf(prow[(long)r * TEN + t] + 1e-8f);
    if (!mk) logp = -__builtin_huge_valf();
    float mx2 = block_reduce_160(logp, red, 0);
    float e   = __expf(logp - mx2);
    float se2 = block_reduce_160(e, red, 1);
    const long o = ((long)b * TDE + m0 + r) * TEN + t;
    out[o]          = e / se2;   // attn
    out[ATTN_N + o] = logp;      // attn_logp
  }
}

// ---------------- launcher ----------------
extern "C" void kernel_launch(void* const* d_in, const int* in_sizes, int n_in,
                              void* d_out, int out_size, void* d_ws, size_t ws_size,
                              hipStream_t stream) {
  const float* queries      = (const float*)d_in[0];
  const float* keys         = (const float*)d_in[1];
  const unsigned char* mask = (const unsigned char*)d_in[2];
  const float* attn_prior   = (const float*)d_in[3];
  const float* kw1 = (const float*)d_in[4];
  const float* kb1 = (const float*)d_in[5];
  const float* kw2 = (const float*)d_in[6];
  const float* kb2 = (const float*)d_in[7];
  const float* qw1 = (const float*)d_in[8];
  const float* qb1 = (const float*)d_in[9];
  const float* qw2 = (const float*)d_in[10];
  const float* qb2 = (const float*)d_in[11];
  const float* qw3 = (const float*)d_in[12];
  const float* qb3 = (const float*)d_in[13];
  float* out = (float*)d_out;

  char* p = (char*)d_ws;
  auto carve = [&](size_t bytes) { char* r = p; p += (bytes + 255) & ~(size_t)255; return r; };
  float*     key_out  = (float*)carve((size_t)B_ * CA * TEN * 4);
  float*     q_out    = (float*)carve((size_t)B_ * CA * TDE * 4);
  _Float16*  kw1p     = (_Float16*)carve((size_t)1024 * 1536 * 2);
  _Float16*  kw2p     = (_Float16*)carve((size_t)80 * 1024 * 2);
  _Float16*  qw1p     = (_Float16*)carve((size_t)160 * 288 * 2);
  _Float16*  qw2p     = (_Float16*)carve((size_t)80 * 160 * 2);
  _Float16*  qw3p     = (_Float16*)carve((size_t)80 * 96 * 2);
  _Float16*  keysT    = (_Float16*)carve((size_t)B_ * 162 * 512 * 2);
  _Float16*  queriesT = (_Float16*)carve((size_t)B_ * 802 * 96 * 2);
  _Float16*  key_outT = (_Float16*)carve((size_t)B_ * TEN * 96 * 2);
  _Float16*  q_outT   = (_Float16*)carve((size_t)B_ * TDE * 96 * 2);

  auto blocks = [](long n) { return (unsigned)((n + 255) / 256); };

  align_pack_w_kernel<<<blocks(1024L * 1536), 256, 0, stream>>>(kw1, kw1p, 1024, 1536, 512, 512, 3, 3, 1);
  align_pack_w_kernel<<<blocks(80L * 1024), 256, 0, stream>>>(kw2, kw2p, 80, 1024, 1024, 1024, 1, 1, 0);
  align_pack_w_kernel<<<blocks(160L * 288), 256, 0, stream>>>(qw1, qw1p, 160, 240, 96, 80, 3, 3, 1);
  align_pack_w_kernel<<<blocks(80L * 160), 256, 0, stream>>>(qw2, qw2p, 80, 160, 160, 160, 1, 1, 0);
  align_pack_w_kernel<<<blocks(80L * 96), 256, 0, stream>>>(qw3, qw3p, 80, 80, 96, 80, 1, 1, 0);
  align_transpose_pad_kernel<<<blocks((long)B_ * 162 * 512), 256, 0, stream>>>(keys, keysT, 512, 160, 512, 162, 1);
  align_transpose_pad_kernel<<<blocks((long)B_ * 802 * 96), 256, 0, stream>>>(queries, queriesT, 80, 800, 96, 802, 1);

  align_key_net_kernel<<<dim3(B_ * 10), dim3(512), 0, stream>>>(kw1p, kb1, kw2p, kb2, keysT, key_out);
  align_query_net_kernel<<<dim3(B_ * 50), dim3(320), 0, stream>>>(qw1p, qb1, qw2p, qb2, qw3p, qb3,
                                                                  queriesT, q_out);
  align_transpose_pad_kernel<<<blocks((long)B_ * TEN * 96), 256, 0, stream>>>(key_out, key_outT, 80, 160, 96, 160, 0);
  align_transpose_pad_kernel<<<blocks((long)B_ * TDE * 96), 256, 0, stream>>>(q_out, q_outT, 80, 800, 96, 800, 0);

  align_attn_kernel<<<dim3(B_ * 50), dim3(160), 0, stream>>>(q_outT, key_outT, q_out, key_out,
                                                             attn_prior, mask, out);

  (void)in_sizes; (void)n_in; (void)out_size; (void)ws_size;
}
